// HeteroSGTModel_31568009625747
// MI455X (gfx1250) — matrix operand order repaired and checked by reading, not compiled, not run on hardware
//
#include <hip/hip_runtime.h>
#include <hip/hip_fp16.h>

typedef __attribute__((ext_vector_type(16))) _Float16 v16h;
typedef __attribute__((ext_vector_type(8)))  float    v8f;
typedef __attribute__((ext_vector_type(4)))  unsigned int v4u;
typedef int v4i_t __attribute__((ext_vector_type(4)));

#define Bc 8
#define Sc 1024
#define Fc 768
#define Ec 512
#define Lc 4
#define Hc 8
#define HDc 64
#define DMAXc 4
#define SCALEc 0.125f

#if defined(__gfx1250__) && __has_builtin(__builtin_amdgcn_global_load_async_to_lds_b128)
#define HAVE_ASYNC 1
#else
#define HAVE_ASYNC 0
#endif

__device__ __forceinline__ v8f wmma_f16(v16h a, v16h b, v8f c) {
  return __builtin_amdgcn_wmma_f32_16x16x32_f16(false, a, false, b, (short)0, c,
                                                false, false);
}

union Frag {
  v16h h;
  unsigned int u[8];
  v4u q[2];
};

// 16-byte global -> LDS copy; async-to-LDS on gfx1250, register bounce otherwise
__device__ __forceinline__ void copy16(const _Float16* g, _Float16* l) {
#if HAVE_ASYNC
  __builtin_amdgcn_global_load_async_to_lds_b128(
      (__attribute__((address_space(1))) v4i_t*)const_cast<_Float16*>(g),
      (__attribute__((address_space(3))) v4i_t*)l, 0, 0);
#else
  *(v4u*)l = *(const v4u*)g;
#endif
}

__device__ __forceinline__ void async_wait0() {
#if HAVE_ASYNC
#if __has_builtin(__builtin_amdgcn_s_wait_asynccnt)
  __builtin_amdgcn_s_wait_asynccnt(0);
#else
  asm volatile("s_wait_asynccnt 0" ::: "memory");
#endif
#endif
}

// ---------------------------------------------------------------------------
// f32 -> f16 cast / transpose-cast preconversion kernels
// ---------------------------------------------------------------------------
__global__ __launch_bounds__(256) void cast16_kernel(
    const float* __restrict__ src, _Float16* __restrict__ dst, int n) {
  int i = blockIdx.x * 256 + threadIdx.x;
  if (i < n) dst[i] = (_Float16)src[i];
}

__global__ __launch_bounds__(256) void transpose16_kernel(
    const float* __restrict__ W, _Float16* __restrict__ Wt, int K, int N) {
  int i = blockIdx.x * 256 + threadIdx.x;
  if (i < K * N) {
    int k = i / N, n = i % N;
    Wt[(size_t)n * K + k] = (_Float16)W[i];
  }
}

// ---------------------------------------------------------------------------
// WMMA GEMM: C[M,N] = A[M,K] (f16 rowmajor) * Wt[N,K]^T (f16) + bias[N]
// MODE 0: f32 row-major to outF
// MODE 1: f16 scattered to (B*H, S, HD)   (Q / K layout)
// MODE 2: f16 scattered to (B*H, HD, S)   (V transposed layout)
// MODE 3: f32 row-major to outF AND f16 row-major to outH
// block = 256 threads (8 waves); tile 128(M) x 64(N); K step 32.
// ---------------------------------------------------------------------------
template <int MODE>
__global__ __launch_bounds__(256) void gemm_wmma(
    const _Float16* __restrict__ A, const _Float16* __restrict__ Wt,
    const float* __restrict__ bias, float* __restrict__ outF,
    _Float16* __restrict__ outH, int M, int N, int K) {
  __shared__ _Float16 As[128 * 32];  // [row][k]
  __shared__ _Float16 Bs[64 * 32];   // [n][k]
  const int tid = threadIdx.x;
  const int wv = tid >> 5, lane = tid & 31;
  const int hl = lane >> 4, ln = lane & 15;
  const int m0 = blockIdx.x * 128;
  const int n0 = blockIdx.y * 64;

  v8f acc[4] = {};
  for (int k0 = 0; k0 < K; k0 += 32) {
    // stage A tile (128x32 halfs) as 16B chunks
#pragma unroll
    for (int c = 0; c < 2; ++c) {
      int cid = tid + 256 * c;            // 0..511
      int r = cid >> 2, off = (cid & 3) << 3;
      copy16(&A[(size_t)(m0 + r) * K + k0 + off], &As[r * 32 + off]);
    }
    // stage Wt tile (64x32 halfs)
    {
      int n = tid >> 2, off = (tid & 3) << 3;  // 0..255
      copy16(&Wt[(size_t)(n0 + n) * K + k0 + off], &Bs[n * 32 + off]);
    }
    async_wait0();
    __syncthreads();
    // A fragment (16x32), contiguous k = 16*hl + s (same order as B)
    Frag af;
    {
      int row = wv * 16 + ln;
      int kb = hl ? 16 : 0;
#pragma unroll
      for (int j = 0; j < 8; ++j) {
        int kk = 2 * j + kb;
        af.u[j] = *(const unsigned int*)&As[row * 32 + kk];
      }
    }
#pragma unroll
    for (int nt = 0; nt < 4; ++nt) {
      Frag bf;
      int n = nt * 16 + ln;
      int kb = hl ? 16 : 0;
#pragma unroll
      for (int j = 0; j < 8; ++j)
        bf.u[j] = *(const unsigned int*)&Bs[n * 32 + kb + 2 * j];
      acc[nt] = wmma_f16(af.h, bf.h, acc[nt]);
    }
    __syncthreads();
  }
  // epilogue: C element (M = r + 8*hl, N = ln)
#pragma unroll
  for (int nt = 0; nt < 4; ++nt) {
    int n = n0 + nt * 16 + ln;
    float bv = bias[n];
#pragma unroll
    for (int r = 0; r < 8; ++r) {
      int m = m0 + wv * 16 + r + 8 * hl;
      float v = acc[nt][r] + bv;
      if (MODE == 0 || MODE == 3) {
        outF[(size_t)m * N + n] = v;
        if (MODE == 3) outH[(size_t)m * N + n] = (_Float16)v;
      } else {
        int b = m >> 10, s = m & (Sc - 1);  // S = 1024
        int h = n >> 6, d = n & (HDc - 1);  // HD = 64
        if (MODE == 1)
          outH[(((size_t)(b * Hc + h) * Sc) + s) * HDc + d] = (_Float16)v;
        else
          outH[(((size_t)(b * Hc + h) * HDc) + d) * Sc + s] = (_Float16)v;
      }
    }
  }
}

// ---------------------------------------------------------------------------
// Flash attention. Block = 4 waves, all on the SAME (b,h); wave w owns q-tile
// (blockIdx&15)*4 + w. K/V tiles for each 32-key step staged once per block.
// q,k: f16 (B*H, S, HD); vT: f16 (B*H, HD, S); o16: f16 (B, S, E).
// ---------------------------------------------------------------------------
__global__ __launch_bounds__(128) void attn_kernel(
    const _Float16* __restrict__ q, const _Float16* __restrict__ k,
    const _Float16* __restrict__ vT, const int* __restrict__ dist,
    const float* __restrict__ dbias_l, _Float16* __restrict__ o16) {
  __shared__ _Float16 Qs[4][16 * 64];
  __shared__ _Float16 Ps[4][16 * 32];
  __shared__ _Float16 Ks[32 * 64];   // [key][d]
  __shared__ _Float16 Vts[64 * 32];  // [d][key]
  __shared__ float db[8];
  const int tid = threadIdx.x;
  const int wv = tid >> 5, lane = tid & 31, hl = lane >> 4, ln = lane & 15;
  if (tid < 8) db[tid] = dbias_l[tid < DMAXc ? tid : DMAXc];

  const int bh = blockIdx.x >> 4;
  const int b = bh >> 3, hh = bh & 7;
  const int q0 = ((blockIdx.x & 15) * 4 + wv) * 16;

  // stage this wave's Q tile (16 x 64 halfs)
#pragma unroll
  for (int c = 0; c < 4; ++c) {
    int cid = lane + 32 * c;              // 0..127
    int r = cid >> 3, off = (cid & 7) << 3;
    copy16(&q[((size_t)bh * Sc + q0 + r) * HDc + off], &Qs[wv][r * 64 + off]);
  }
  async_wait0();
  __syncthreads();

  Frag aq0, aq1;  // A fragments for d=0..31 and d=32..63
  {
    int row = ln;
    int kb = hl ? 16 : 0;
#pragma unroll
    for (int j = 0; j < 8; ++j) {
      int kk = 2 * j + kb;
      aq0.u[j] = *(const unsigned int*)&Qs[wv][row * 64 + kk];
      aq1.u[j] = *(const unsigned int*)&Qs[wv][row * 64 + 32 + kk];
    }
  }

  v8f oa[4] = {};
  float mrow[8], lsum[8];
#pragma unroll
  for (int r = 0; r < 8; ++r) {
    mrow[r] = -3.0e38f;
    lsum[r] = 0.0f;
  }
  const size_t kbase = (size_t)bh * Sc * HDc;
  const size_t vbase = (size_t)bh * HDc * Sc;

  for (int j0 = 0; j0 < Sc; j0 += 32) {
    __syncthreads();  // previous step's Ks/Vts reads complete
    // stage K tile (32 keys x 64 d) and Vt tile (64 d x 32 keys)
#pragma unroll
    for (int c = 0; c < 2; ++c) {
      int cid = tid + 128 * c;            // 0..255
      int r = cid >> 3, off = (cid & 7) << 3;
      copy16(&k[kbase + (size_t)(j0 + r) * HDc + off], &Ks[r * 64 + off]);
      int n = cid >> 2, voff = (cid & 3) << 3;
      copy16(&vT[vbase + (size_t)n * Sc + j0 + voff], &Vts[n * 32 + voff]);
    }
    async_wait0();
    __syncthreads();

    v8f s0 = {}, s1 = {};
    {
      int dbse = hl ? 16 : 0;
      Frag bk;
      // tile 0: keys j0..j0+15
#pragma unroll
      for (int j = 0; j < 8; ++j)
        bk.u[j] = *(const unsigned int*)&Ks[ln * 64 + dbse + 2 * j];
      s0 = wmma_f16(aq0.h, bk.h, s0);
#pragma unroll
      for (int j = 0; j < 8; ++j)
        bk.u[j] = *(const unsigned int*)&Ks[ln * 64 + 32 + dbse + 2 * j];
      s0 = wmma_f16(aq1.h, bk.h, s0);
      // tile 1: keys j0+16..j0+31
#pragma unroll
      for (int j = 0; j < 8; ++j)
        bk.u[j] = *(const unsigned int*)&Ks[(16 + ln) * 64 + dbse + 2 * j];
      s1 = wmma_f16(aq0.h, bk.h, s1);
#pragma unroll
      for (int j = 0; j < 8; ++j)
        bk.u[j] = *(const unsigned int*)&Ks[(16 + ln) * 64 + 32 + dbse + 2 * j];
      s1 = wmma_f16(aq1.h, bk.h, s1);
    }
    // scale + distance bias
#pragma unroll
    for (int r = 0; r < 8; ++r) {
      int qi = q0 + r + 8 * hl;
      const int* dp = &dist[((size_t)b * Sc + qi) * Sc + j0];
      int d0 = dp[ln];
      int d1 = dp[16 + ln];
      d0 = d0 < 0 ? 0 : (d0 > 7 ? 7 : d0);
      d1 = d1 < 0 ? 0 : (d1 > 7 ? 7 : d1);
      s0[r] = s0[r] * SCALEc + db[d0];
      s1[r] = s1[r] * SCALEc + db[d1];
    }
    // online softmax over these 32 columns
#pragma unroll
    for (int r = 0; r < 8; ++r) {
      float mt = fmaxf(s0[r], s1[r]);
      mt = fmaxf(mt, __shfl_xor(mt, 1));
      mt = fmaxf(mt, __shfl_xor(mt, 2));
      mt = fmaxf(mt, __shfl_xor(mt, 4));
      mt = fmaxf(mt, __shfl_xor(mt, 8));
      float mnew = fmaxf(mrow[r], mt);
      float scl = __expf(mrow[r] - mnew);
      float p0 = __expf(s0[r] - mnew);
      float p1 = __expf(s1[r] - mnew);
      s0[r] = p0;
      s1[r] = p1;
      float ps = p0 + p1;
      ps += __shfl_xor(ps, 1);
      ps += __shfl_xor(ps, 2);
      ps += __shfl_xor(ps, 4);
      ps += __shfl_xor(ps, 8);
      lsum[r] = lsum[r] * scl + ps;
      mrow[r] = mnew;
      oa[0][r] *= scl; oa[1][r] *= scl; oa[2][r] *= scl; oa[3][r] *= scl;
    }
    // P (16x32) through LDS into A-fragment layout
#pragma unroll
    for (int r = 0; r < 8; ++r) {
      int row = r + 8 * hl;
      Ps[wv][row * 32 + ln] = (_Float16)s0[r];
      Ps[wv][row * 32 + 16 + ln] = (_Float16)s1[r];
    }
    __syncthreads();
    Frag ap;
    {
      int row = ln;
      int kb = hl ? 16 : 0;
#pragma unroll
      for (int j = 0; j < 8; ++j) {
        int kk = 2 * j + kb;
        ap.u[j] = *(const unsigned int*)&Ps[wv][row * 32 + kk];
      }
    }
    // P(16x32) x V(32x64)
    {
      int koff = hl ? 16 : 0;
#pragma unroll
      for (int nt = 0; nt < 4; ++nt) {
        Frag bv;
        int n = nt * 16 + ln;
#pragma unroll
        for (int j = 0; j < 8; ++j)
          bv.u[j] = *(const unsigned int*)&Vts[n * 32 + koff + 2 * j];
        oa[nt] = wmma_f16(ap.h, bv.h, oa[nt]);
      }
    }
  }
  // normalize + store f16 o (B, S, E) with head interleave
#pragma unroll
  for (int r = 0; r < 8; ++r) {
    float inv = 1.0f / lsum[r];
    int s = q0 + r + 8 * hl;
    size_t base = ((size_t)b * Sc + s) * Ec + hh * HDc;
#pragma unroll
    for (int nt = 0; nt < 4; ++nt) {
      o16[base + nt * 16 + ln] = (_Float16)(oa[nt][r] * inv);
    }
  }
}

// ---------------------------------------------------------------------------
// h = LayerNorm(h + tmp) * g + b  (f32, in place) ; also emits f16 copy h16
// ---------------------------------------------------------------------------
__global__ __launch_bounds__(256) void ln_kernel(
    float* __restrict__ h, const float* __restrict__ tmp,
    const float* __restrict__ g, const float* __restrict__ bb,
    _Float16* __restrict__ h16) {
  const int wv = threadIdx.x >> 5, lane = threadIdx.x & 31;
  const int row = blockIdx.x * 8 + wv;
  float vals[16];
  const float* hr = &h[(size_t)row * Ec];
  const float* tr = &tmp[(size_t)row * Ec];
  float sum = 0.0f;
#pragma unroll
  for (int i = 0; i < 16; ++i) {
    int e = lane + 32 * i;
    vals[i] = hr[e] + tr[e];
    sum += vals[i];
  }
#pragma unroll
  for (int m = 1; m < 32; m <<= 1) sum += __shfl_xor(sum, m);
  float mean = sum * (1.0f / Ec);
  float vs = 0.0f;
#pragma unroll
  for (int i = 0; i < 16; ++i) {
    float d = vals[i] - mean;
    vs += d * d;
  }
#pragma unroll
  for (int m = 1; m < 32; m <<= 1) vs += __shfl_xor(vs, m);
  float rs = rsqrtf(vs * (1.0f / Ec) + 1e-5f);
#pragma unroll
  for (int i = 0; i < 16; ++i) {
    int e = lane + 32 * i;
    float y = (vals[i] - mean) * rs * g[e] + bb[e];
    h[(size_t)row * Ec + e] = y;
    h16[(size_t)row * Ec + e] = (_Float16)y;
  }
}

// ---------------------------------------------------------------------------
// head: t = h[:,0]; hid = relu(t@W1+b1) (8x256); logits = hid@W2+b2 (8x2)
// ---------------------------------------------------------------------------
__global__ __launch_bounds__(256) void head_kernel(
    const float* __restrict__ h, const float* __restrict__ W1,
    const float* __restrict__ b1, const float* __restrict__ W2,
    const float* __restrict__ b2, float* __restrict__ out) {
  __shared__ float hid[8][256];
  const int j = threadIdx.x;
  for (int b = 0; b < 8; ++b) {
    const float* t = &h[(size_t)b * Sc * Ec];
    float acc = b1[j];
    for (int e = 0; e < Ec; ++e) acc += t[e] * W1[(size_t)e * 256 + j];
    hid[b][j] = fmaxf(acc, 0.0f);
  }
  __syncthreads();
  if (j < 16) {
    int b = j >> 1, c = j & 1;
    float acc = b2[c];
    for (int e = 0; e < 256; ++e) acc += hid[b][e] * W2[e * 2 + c];
    out[b * 2 + c] = acc;
  }
}

// ---------------------------------------------------------------------------
extern "C" void kernel_launch(void* const* d_in, const int* in_sizes, int n_in,
                              void* d_out, int out_size, void* d_ws,
                              size_t ws_size, hipStream_t stream) {
  const float* x     = (const float*)d_in[0];
  const int*   dist  = (const int*)d_in[1];
  const float* Win   = (const float*)d_in[2];
  const float* b_in  = (const float*)d_in[3];
  const float* Wq    = (const float*)d_in[4];
  const float* bq    = (const float*)d_in[5];
  const float* Wk    = (const float*)d_in[6];
  const float* bk    = (const float*)d_in[7];
  const float* Wv    = (const float*)d_in[8];
  const float* bv    = (const float*)d_in[9];
  const float* Wo    = (const float*)d_in[10];
  const float* bo    = (const float*)d_in[11];
  const float* dbias = (const float*)d_in[12];
  const float* ln_g  = (const float*)d_in[13];
  const float* ln_b  = (const float*)d_in[14];
  const float* W1    = (const float*)d_in[15];
  const float* b1    = (const float*)d_in[16];
  const float* W2    = (const float*)d_in[17];
  const float* b2    = (const float*)d_in[18];
  float* out = (float*)d_out;

  char* ws = (char*)d_ws;
  size_t off = 0;
  auto carve = [&](size_t bytes) {
    void* p = ws + off;
    off += (bytes + 255) & ~(size_t)255;
    return p;
  };
  const size_t MR = (size_t)Bc * Sc;  // 8192 rows
  float*    h    = (float*)carve(MR * Ec * sizeof(float));
  float*    tmp  = (float*)carve(MR * Ec * sizeof(float));
  _Float16* h16  = (_Float16*)carve(MR * Ec * sizeof(_Float16));
  _Float16* o16  = (_Float16*)carve(MR * Ec * sizeof(_Float16));
  _Float16* q16  = (_Float16*)carve(MR * Ec * sizeof(_Float16));
  _Float16* k16  = (_Float16*)carve(MR * Ec * sizeof(_Float16));
  _Float16* vT16 = (_Float16*)carve(MR * Ec * sizeof(_Float16));
  _Float16* x16  = (_Float16*)carve(MR * Fc * sizeof(_Float16));
  _Float16* WinT = (_Float16*)carve((size_t)Fc * Ec * sizeof(_Float16));
  _Float16* WqT  = (_Float16*)carve((size_t)Lc * Ec * Ec * sizeof(_Float16));
  _Float16* WkT  = (_Float16*)carve((size_t)Lc * Ec * Ec * sizeof(_Float16));
  _Float16* WvT  = (_Float16*)carve((size_t)Lc * Ec * Ec * sizeof(_Float16));
  _Float16* WoT  = (_Float16*)carve((size_t)Lc * Ec * Ec * sizeof(_Float16));

  // ---- preconversion: x and all weights to f16 (weights transposed N x K)
  cast16_kernel<<<(int)((MR * Fc + 255) / 256), 256, 0, stream>>>(
      x, x16, (int)(MR * Fc));
  transpose16_kernel<<<(Fc * Ec + 255) / 256, 256, 0, stream>>>(Win, WinT, Fc,
                                                                Ec);
  for (int l = 0; l < Lc; ++l) {
    size_t wo = (size_t)l * Ec * Ec;
    transpose16_kernel<<<(Ec * Ec + 255) / 256, 256, 0, stream>>>(
        Wq + wo, WqT + wo, Ec, Ec);
    transpose16_kernel<<<(Ec * Ec + 255) / 256, 256, 0, stream>>>(
        Wk + wo, WkT + wo, Ec, Ec);
    transpose16_kernel<<<(Ec * Ec + 255) / 256, 256, 0, stream>>>(
        Wv + wo, WvT + wo, Ec, Ec);
    transpose16_kernel<<<(Ec * Ec + 255) / 256, 256, 0, stream>>>(
        Wo + wo, WoT + wo, Ec, Ec);
  }

  dim3 gp(MR / 128, Ec / 64);  // (64, 8)

  // input projection: h (f32) + h16 (f16) = x16 @ WinT^T + b_in
  gemm_wmma<3><<<gp, 256, 0, stream>>>(x16, WinT, b_in, h, h16, (int)MR, Ec,
                                       Fc);
  for (int l = 0; l < Lc; ++l) {
    size_t wo = (size_t)l * Ec * Ec;
    gemm_wmma<1><<<gp, 256, 0, stream>>>(h16, WqT + wo, bq + l * Ec, nullptr,
                                         q16, (int)MR, Ec, Ec);
    gemm_wmma<1><<<gp, 256, 0, stream>>>(h16, WkT + wo, bk + l * Ec, nullptr,
                                         k16, (int)MR, Ec, Ec);
    gemm_wmma<2><<<gp, 256, 0, stream>>>(h16, WvT + wo, bv + l * Ec, nullptr,
                                         vT16, (int)MR, Ec, Ec);
    attn_kernel<<<Bc * Hc * (Sc / 64), 128, 0, stream>>>(
        q16, k16, vT16, dist, dbias + l * (DMAXc + 1), o16);
    gemm_wmma<0><<<gp, 256, 0, stream>>>(o16, WoT + wo, bo + l * Ec, tmp,
                                         nullptr, (int)MR, Ec, Ec);
    ln_kernel<<<MR / 8, 256, 0, stream>>>(h, tmp, ln_g + (size_t)l * Ec,
                                          ln_b + (size_t)l * Ec, h16);
  }
  head_kernel<<<1, 256, 0, stream>>>(h, W1, b1, W2, b2, out);
}
